// MultiHeadAttention_15015205667536
// MI455X (gfx1250) — compile-verified, matrix operand order checked by reading, not run on hardware
//
#include <hip/hip_runtime.h>
#include <hip/hip_bf16.h>

typedef _Float16 f16;
typedef __attribute__((ext_vector_type(16))) _Float16 v16h;
typedef __attribute__((ext_vector_type(8)))  _Float16 v8h;
typedef __attribute__((ext_vector_type(4)))  _Float16 v4h;
typedef __attribute__((ext_vector_type(8)))  float    v8f;
typedef __attribute__((ext_vector_type(4)))  float    v4f;

#define DM    1024
#define HEADS 16
#define HD    64
#define SEQ   2048
#define BATCH 4

// ---------------- elementwise f32 -> f16 convert (vectorized x4) ----------------
__global__ void __launch_bounds__(256) cvt_f32_f16(const float* __restrict__ x,
                                                   f16* __restrict__ y, long n4) {
  long i = (long)blockIdx.x * 256 + threadIdx.x;
  if (i < n4) {
    v4f v = *((const v4f*)x + i);
    v4h o;
#pragma unroll
    for (int e = 0; e < 4; e++) o[e] = (f16)v[e];
    *((v4h*)y + i) = o;
  }
}

// ---------------- V head transpose: vp16[B,S,D] -> vt[b,h,d,s] (K=s contiguous) ----------------
__global__ void __launch_bounds__(256) vtrans(const f16* __restrict__ vp,
                                              f16* __restrict__ vt) {
  long idx = (long)blockIdx.x * 256 + threadIdx.x;  // over B*H*HD*SEQ
  int  s   = (int)(idx & (SEQ - 1));
  long r1  = idx >> 11;
  int  d   = (int)(r1 & (HD - 1));
  long r2  = r1 >> 6;
  int  h   = (int)(r2 & (HEADS - 1));
  int  b   = (int)(r2 >> 4);
  vt[idx] = vp[((long)(b * SEQ + s)) * DM + h * HD + d];
}

// ---------------- row softmax, in place, S=2048, block=256 (8 elems/thread) ----------------
__global__ void __launch_bounds__(256) softmax_rows(float* __restrict__ attn) {
  const int tid = threadIdx.x;
  float* p = attn + (long)blockIdx.x * SEQ;
  float v[8];
  float mx = -3.4e38f;
#pragma unroll
  for (int e = 0; e < 8; e++) { v[e] = p[tid + e * 256]; mx = fmaxf(mx, v[e]); }
  __shared__ float red[256];
  red[tid] = mx; __syncthreads();
  for (int s = 128; s > 0; s >>= 1) { if (tid < s) red[tid] = fmaxf(red[tid], red[tid + s]); __syncthreads(); }
  mx = red[0]; __syncthreads();
  float sum = 0.f;
#pragma unroll
  for (int e = 0; e < 8; e++) { v[e] = __expf(v[e] - mx); sum += v[e]; }
  red[tid] = sum; __syncthreads();
  for (int s = 128; s > 0; s >>= 1) { if (tid < s) red[tid] += red[tid + s]; __syncthreads(); }
  float inv = 1.0f / red[0];
#pragma unroll
  for (int e = 0; e < 8; e++) p[tid + e * 256] = v[e] * inv;
}

// ---------------- generic NT WMMA GEMM ----------------
// C[m,n] = scale * sum_k A[m,k]*B[n,k] + bias[n], optional mask (==0 -> -1e10 after scale).
// A: f16 or f32 (A_F32), converted to f16 fragments. B: f16 [N,K] row-major.
// C store: f16 or f32 (C_F16). Batched via blockIdx.z = zb*H + zh with element strides.
// Wave tile: 64x64, K-step 32 -> 16 x v_wmma_f32_16x16x32_f16 per step.
// Fragment layout (wave32): lane = row index (0..15, dup in 16..31), per-lane halves
// e0..7 at K = kg..kg+7, e8..15 at K = kg+16..kg+23, kg = (lane>=16)*8.
template <bool A_F32, bool C_F16>
__global__ void __launch_bounds__(256)
gemm_nt_wmma(const void* __restrict__ Av, const f16* __restrict__ Bm,
             void* __restrict__ Cv, const float* __restrict__ bias,
             const int* __restrict__ mask, float scale,
             int M, int N, int K, int lda, int ldb, int ldc,
             long sAb, long sAh, long sBb, long sBh, long sCb, long sCh,
             long sMb, int H) {
  const int z    = blockIdx.z;
  const int zb   = z / H, zh = z - zb * H;
  const int lane = threadIdx.x;          // 0..31
  const int wave = threadIdx.y;          // 0..7
  const int n0   = blockIdx.x * 64;
  const int m0   = (blockIdx.y * 8 + wave) * 64;
  if (m0 >= M || n0 >= N) return;

  const int r  = lane & 15;
  const int kg = (lane >> 4) << 3;  // 0 or 8

  const f16* B = Bm + zb * sBb + zh * sBh;
  const f16*   A16 = nullptr;
  const float* A32 = nullptr;
  if constexpr (A_F32) A32 = (const float*)Av + zb * sAb + zh * sAh;
  else                 A16 = (const f16*)Av + zb * sAb + zh * sAh;

  const f16*   bp[4];
  const f16*   ap16[4];
  const float* ap32[4];
#pragma unroll
  for (int i = 0; i < 4; i++) {
    bp[i] = B + (long)(n0 + i * 16 + r) * ldb + kg;
    if constexpr (A_F32) ap32[i] = A32 + (long)(m0 + i * 16 + r) * lda + kg;
    else                 ap16[i] = A16 + (long)(m0 + i * 16 + r) * lda + kg;
  }

  v8f acc[4][4] = {};

  for (int kb = 0; kb < K; kb += 32) {
    if (kb + 32 < K) {
      __builtin_prefetch(bp[0] + kb + 32, 0, 1);
      if constexpr (A_F32) __builtin_prefetch(ap32[0] + kb + 32, 0, 1);
      else                 __builtin_prefetch(ap16[0] + kb + 32, 0, 1);
    }
    v16h af[4], bf[4];
#pragma unroll
    for (int i = 0; i < 4; i++) {
      if constexpr (A_F32) {
        const float* p = ap32[i] + kb;
        v4f f0 = *(const v4f*)p;
        v4f f1 = *(const v4f*)(p + 4);
        v4f f2 = *(const v4f*)(p + 16);
        v4f f3 = *(const v4f*)(p + 20);
        v16h a;
#pragma unroll
        for (int e = 0; e < 4; e++) {
          a[e] = (f16)f0[e]; a[e + 4] = (f16)f1[e];
          a[e + 8] = (f16)f2[e]; a[e + 12] = (f16)f3[e];
        }
        af[i] = a;
      } else {
        const f16* p = ap16[i] + kb;
        v8h lo = *(const v8h*)p;
        v8h hi = *(const v8h*)(p + 16);
        af[i] = __builtin_shufflevector(lo, hi, 0, 1, 2, 3, 4, 5, 6, 7,
                                        8, 9, 10, 11, 12, 13, 14, 15);
      }
    }
#pragma unroll
    for (int j = 0; j < 4; j++) {
      const f16* p = bp[j] + kb;
      v8h lo = *(const v8h*)p;
      v8h hi = *(const v8h*)(p + 16);
      bf[j] = __builtin_shufflevector(lo, hi, 0, 1, 2, 3, 4, 5, 6, 7,
                                      8, 9, 10, 11, 12, 13, 14, 15);
    }
#pragma unroll
    for (int i = 0; i < 4; i++)
#pragma unroll
      for (int j = 0; j < 4; j++)
        acc[i][j] = __builtin_amdgcn_wmma_f32_16x16x32_f16(
            false, af[i], false, bf[j], (short)0, acc[i][j], false, false);
  }

  // epilogue: C VGPR t holds M = t (lanes 0-15) / t+8 (lanes 16-31), N = lane&15
  float* C32 = nullptr; f16* C16 = nullptr;
  const long cOff = zb * sCb + zh * sCh;
  if constexpr (C_F16) C16 = (f16*)Cv + cOff; else C32 = (float*)Cv + cOff;
  const int* mrow = mask ? (mask + zb * sMb) : nullptr;
  const int  mHi  = (lane >> 4) << 3;

#pragma unroll
  for (int j = 0; j < 4; j++) {
    const int n = n0 + j * 16 + r;
    const float bv = bias ? bias[n] : 0.0f;
#pragma unroll
    for (int i = 0; i < 4; i++) {
#pragma unroll
      for (int t = 0; t < 8; t++) {
        const int m = m0 + i * 16 + mHi + t;
        float val = acc[i][j][t] * scale + bv;
        if (mrow && mrow[(long)m * N + n] == 0) val = -1e10f;
        const long ci = (long)m * ldc + n;
        if constexpr (C_F16) C16[ci] = (f16)val; else C32[ci] = val;
      }
    }
  }
}

// ---------------- host launch ----------------
extern "C" void kernel_launch(void* const* d_in, const int* in_sizes, int n_in,
                              void* d_out, int out_size, void* d_ws, size_t ws_size,
                              hipStream_t stream) {
  (void)in_sizes; (void)n_in; (void)out_size; (void)ws_size;
  const float* q  = (const float*)d_in[0];
  const float* k  = (const float*)d_in[1];
  const float* v  = (const float*)d_in[2];
  const int*  msk = (const int*)d_in[3];
  const float* wq = (const float*)d_in[4];
  const float* bq = (const float*)d_in[5];
  const float* wk = (const float*)d_in[6];
  const float* bk = (const float*)d_in[7];
  const float* wv = (const float*)d_in[8];
  const float* bv = (const float*)d_in[9];
  const float* wo = (const float*)d_in[10];
  const float* bo = (const float*)d_in[11];

  const long MX  = (long)BATCH * SEQ;  // 8192 rows
  const long NXD = MX * DM;            // 8,388,608
  const long NW  = (long)DM * DM;      // 1,048,576

  f16* w = (f16*)d_ws;
  f16* q16  = w; w += NXD;
  f16* k16  = w; w += NXD;
  f16* v16  = w; w += NXD;
  f16* wq16 = w; w += NW;
  f16* wk16 = w; w += NW;
  f16* wv16 = w; w += NW;
  f16* wo16 = w; w += NW;
  f16* qp16 = w; w += NXD;
  f16* kp16 = w; w += NXD;
  f16* vp16 = w; w += NXD;
  f16* vt16 = w; w += NXD;
  f16* ctx16 = w; w += NXD;

  float* outp = (float*)d_out;
  float* attn = outp + NXD;  // attn region of d_out, also used as scores scratch

  // 1) convert inputs + weights to f16
  cvt_f32_f16<<<(unsigned)(NXD / 4 / 256), 256, 0, stream>>>(q, q16, NXD / 4);
  cvt_f32_f16<<<(unsigned)(NXD / 4 / 256), 256, 0, stream>>>(k, k16, NXD / 4);
  cvt_f32_f16<<<(unsigned)(NXD / 4 / 256), 256, 0, stream>>>(v, v16, NXD / 4);
  cvt_f32_f16<<<(unsigned)(NW / 4 / 256), 256, 0, stream>>>(wq, wq16, NW / 4);
  cvt_f32_f16<<<(unsigned)(NW / 4 / 256), 256, 0, stream>>>(wk, wk16, NW / 4);
  cvt_f32_f16<<<(unsigned)(NW / 4 / 256), 256, 0, stream>>>(wv, wv16, NW / 4);
  cvt_f32_f16<<<(unsigned)(NW / 4 / 256), 256, 0, stream>>>(wo, wo16, NW / 4);

  const dim3 gblk(32, 8, 1);

  // 2) projections: X[8192,1024] @ W[1024,1024]^T + b -> f16
  gemm_nt_wmma<false, true><<<dim3(DM / 64, (unsigned)(MX / 512), 1), gblk, 0, stream>>>(
      q16, wq16, qp16, bq, nullptr, 1.0f, (int)MX, DM, DM, DM, DM, DM,
      0, 0, 0, 0, 0, 0, 0, 1);
  gemm_nt_wmma<false, true><<<dim3(DM / 64, (unsigned)(MX / 512), 1), gblk, 0, stream>>>(
      k16, wk16, kp16, bk, nullptr, 1.0f, (int)MX, DM, DM, DM, DM, DM,
      0, 0, 0, 0, 0, 0, 0, 1);
  gemm_nt_wmma<false, true><<<dim3(DM / 64, (unsigned)(MX / 512), 1), gblk, 0, stream>>>(
      v16, wv16, vp16, bv, nullptr, 1.0f, (int)MX, DM, DM, DM, DM, DM,
      0, 0, 0, 0, 0, 0, 0, 1);

  // 3) V head transpose for K-contiguous B operand
  vtrans<<<(unsigned)(NXD / 256), 256, 0, stream>>>(vp16, vt16);

  // 4) scores = (Qh Kh^T)/8, masked, fp32 -> attn region of d_out
  gemm_nt_wmma<false, false><<<dim3(SEQ / 64, SEQ / 512, BATCH * HEADS), gblk, 0, stream>>>(
      qp16, kp16, attn, nullptr, msk, 0.125f,
      SEQ, SEQ, HD, DM, DM, SEQ,
      (long)SEQ * DM, (long)HD,            // A: batch, head strides
      (long)SEQ * DM, (long)HD,            // B
      (long)HEADS * SEQ * SEQ, (long)SEQ * SEQ,  // C
      (long)SEQ * SEQ, HEADS);             // mask batch stride

  // 5) softmax rows in place (mask already folded in as -1e10)
  softmax_rows<<<(unsigned)(BATCH * HEADS * SEQ), 256, 0, stream>>>(attn);

  // 6) ctx = attn @ Vh  (A fp32 from d_out, inline cvt; C f16 merged heads)
  gemm_nt_wmma<true, true><<<dim3(1, SEQ / 512, BATCH * HEADS), gblk, 0, stream>>>(
      attn, vt16, ctx16, nullptr, nullptr, 1.0f,
      SEQ, HD, SEQ, SEQ, SEQ, DM,
      (long)HEADS * SEQ * SEQ, (long)SEQ * SEQ,  // A
      (long)DM * SEQ, (long)HD * SEQ,            // B (Vt)
      (long)SEQ * DM, (long)HD,                  // C (ctx, col offset h*64)
      0, HEADS);

  // 7) out = ctx @ Wo^T + bo -> fp32 out region
  gemm_nt_wmma<false, false><<<dim3(DM / 64, (unsigned)(MX / 512), 1), gblk, 0, stream>>>(
      ctx16, wo16, outp, bo, nullptr, 1.0f, (int)MX, DM, DM, DM, DM, DM,
      0, 0, 0, 0, 0, 0, 0, 1);
}
